// AGCRN_89696097010338
// MI455X (gfx1250) — matrix operand order repaired and checked by reading, not compiled, not run on hardware
//
#include <hip/hip_runtime.h>
#include <cstdint>
#include <cstddef>

// ---- problem constants ----
#define NN   1024
#define BB   32
#define TT   12
#define HH   64
#define DD   16
#define OUTC 12
// layer0: in=66 (C=2 + H=64), KI=132 -> pad 160 ; layer1: in=128, KI=256

typedef __bf16 v16bf __attribute__((ext_vector_type(16)));
typedef __bf16 v8bf  __attribute__((ext_vector_type(8)));
typedef float  v8f   __attribute__((ext_vector_type(8)));
typedef int    v4i_t __attribute__((vector_size(4 * sizeof(int))));

#define AS1 __attribute__((address_space(1)))
#define AS3 __attribute__((address_space(3)))

// ---------------- async global -> LDS staging (CDNA5 ASYNCcnt path) ----------------
// 16 bytes per lane. Falls back to a synchronous copy if the builtin is absent.
__device__ inline void async_copy16(void* lds, const void* gsrc) {
#if defined(__AMDGCN__) && __has_builtin(__builtin_amdgcn_global_load_async_to_lds_b128)
  __builtin_amdgcn_global_load_async_to_lds_b128((AS1 v4i_t*)gsrc, (AS3 v4i_t*)lds, 0, 0);
#else
  *(uint4*)lds = *(const uint4*)gsrc;
#endif
}
__device__ inline void async_wait_all() {
#if defined(__AMDGCN__) && __has_builtin(__builtin_amdgcn_s_wait_asynccnt)
  __builtin_amdgcn_s_wait_asynccnt(0);
#elif defined(__AMDGCN__)
  asm volatile("s_wait_asynccnt 0x0" ::: "memory");
#endif
}

// ---------------- WMMA helpers (wave32, gfx1250) ----------------
__device__ inline v8f wmma_bf16(v16bf a, v16bf b, v8f c) {
  // D = A(16x32 bf16) * B(32x16 bf16) + C(16x16 f32)
  return __builtin_amdgcn_wmma_f32_16x16x32_bf16(false, a, false, b, (short)0, c, false, false);
}

// A-fragment from ROW-MAJOR LDS tile (16x32, row stride in elements).
// lane (m = lane%16, g = lane/16): h0..7 = row m cols [8g,8g+8),
// h8..15 = row m cols [16+8g,16+8g+8)  -> two 16B LDS loads.
__device__ inline v16bf lds_afrag(const __bf16* p, int stride) {
  const int lane = threadIdx.x & 31;
  const int m = lane & 15;
  const int g = (lane >> 4) & 1;
  const __bf16* q = p + m * stride + g * 8;
  v8bf lo = *(const v8bf*)q;
  v8bf hi = *(const v8bf*)(q + 16);
  return __builtin_shufflevector(lo, hi, 0, 1, 2, 3, 4, 5, 6, 7,
                                         8, 9, 10, 11, 12, 13, 14, 15);
}

// B-fragment from TRANSPOSED LDS tile T[col][k], k-stride 32.
// element (k, n) at T + n*32 + k ; lane (nc,g) needs k = [8g,8g+8) then [16+8g,16+8g+8).
__device__ inline v16bf lds_bfragT(const __bf16* p) {
  const int lane = threadIdx.x & 31;
  const int nc = lane & 15;
  const int g = (lane >> 4) & 1;
  const __bf16* q = p + nc * 32 + g * 8;
  v8bf lo = *(const v8bf*)q;
  v8bf hi = *(const v8bf*)(q + 16);
  return __builtin_shufflevector(lo, hi, 0, 1, 2, 3, 4, 5, 6, 7,
                                         8, 9, 10, 11, 12, 13, 14, 15);
}
// C/D f32 16x16: vgpr j -> row m = j + 8*(lane/16), col n = lane%16

// ---------------- A = softmax(relu(E E^T), axis=1), bf16 ----------------
__global__ __launch_bounds__(256) void agcrn_build_A(const float* __restrict__ emb,
                                                     __bf16* __restrict__ Abf) {
  __shared__ float sh[NN];
  __shared__ float red[256];
  const int n   = blockIdx.x;
  const int tid = threadIdx.x;
  const float* en = emb + n * DD;

  float lmax = 0.0f;  // relu output >= 0
  for (int m = tid; m < NN; m += 256) {
    float d = 0.0f;
#pragma unroll
    for (int dd = 0; dd < DD; ++dd) d += en[dd] * emb[m * DD + dd];
    float r = fmaxf(d, 0.0f);
    sh[m] = r;
    lmax = fmaxf(lmax, r);
  }
  red[tid] = lmax; __syncthreads();
  for (int s = 128; s > 0; s >>= 1) { if (tid < s) red[tid] = fmaxf(red[tid], red[tid + s]); __syncthreads(); }
  const float mx = red[0];
  __syncthreads();

  float ls = 0.0f;
  for (int m = tid; m < NN; m += 256) {
    float e = __expf(sh[m] - mx);
    sh[m] = e;
    ls += e;
  }
  red[tid] = ls; __syncthreads();
  for (int s = 128; s > 0; s >>= 1) { if (tid < s) red[tid] += red[tid + s]; __syncthreads(); }
  const float inv = 1.0f / red[0];
  __syncthreads();

  for (int m = tid; m < NN; m += 256)
    Abf[(unsigned)(n * NN + m)] = (__bf16)(sh[m] * inv);
}

// ---------------- per-node weights, TRANSPOSED: Wn[n][o][j] = sum_d emb[n][d]*wp[d][k][i][o] ----
// j = k*I + i ; zero for j in [2I, KIP)
__global__ __launch_bounds__(256) void agcrn_build_w(const float* __restrict__ emb,
                                                     const float* __restrict__ wp,
                                                     __bf16* __restrict__ Wn,
                                                     int I, int KIP, int O) {
  const int n   = blockIdx.y;
  const int idx = blockIdx.x * 256 + threadIdx.x;
  if (idx >= KIP * O) return;
  const int j = idx / O;
  const int o = idx - j * O;
  float acc = 0.0f;
  if (j < 2 * I) {
    const int k = (j >= I) ? 1 : 0;
    const int i = j - k * I;
    const float* base = wp + (unsigned)((k * I + i) * O + o);
    const unsigned step = (unsigned)(2 * I * O);
#pragma unroll
    for (int d = 0; d < DD; ++d) acc += emb[n * DD + d] * base[(unsigned)d * step];
  }
  Wn[((unsigned)n * O + o) * (unsigned)KIP + j] = (__bf16)acc;   // transposed [n][o][j]
}

__global__ void agcrn_build_b(const float* __restrict__ emb,
                              const float* __restrict__ bp,
                              float* __restrict__ Bn, int O) {
  const int n = blockIdx.x;
  const int o = threadIdx.x;
  if (o >= O) return;
  float acc = 0.0f;
#pragma unroll
  for (int d = 0; d < DD; ++d) acc += emb[n * DD + d] * bp[d * O + o];
  Bn[(unsigned)(n * O + o)] = acc;
}

// ---------------- pack: concat input in two layouts (bf16) ----------------
// aggin[b][i][m]  (X^T per batch, aggregation-GEMM B-operand staging)
// xg[m][b][j=i]   (identity support part, A-operand of per-node GEMM)
// one block per (b,m): no integer division anywhere
__global__ __launch_bounds__(128) void agcrn_pack(const float* __restrict__ x,
                                                  int xbs, int xns, int Cx,
                                                  const float* __restrict__ h,
                                                  const float* __restrict__ zr,
                                                  __bf16* __restrict__ aggin,
                                                  __bf16* __restrict__ xg,
                                                  int I, int KIP) {
  const unsigned bm = blockIdx.x;          // [0, BB*NN)
  const unsigned m  = bm & (NN - 1);
  const unsigned b  = bm >> 10;
  const unsigned hbase = (b * NN + m) * HH;
  const unsigned zbase = (b * NN + m) * 128;
  for (unsigned i = threadIdx.x; i < (unsigned)I; i += 128) {
    float v;
    if (i < (unsigned)Cx) {
      v = x[b * (unsigned)xbs + m * (unsigned)xns + i];
    } else {
      v = h[hbase + (i - Cx)];
      if (zr) v *= zr[zbase + (i - Cx)];   // z = first half of zr
    }
    const __bf16 bv = (__bf16)v;
    aggin[(b * I + i) * NN + m] = bv;      // transposed [b][i][m]
    xg[(m * BB + b) * (unsigned)KIP + i] = bv;
  }
}

// ---------------- GEMM1: Y[b] = A(1024x1024) @ X[b](1024xI), write bf16 into xg[j=I+i] ----------------
__global__ __launch_bounds__(128) void agcrn_gemm_agg(const __bf16* __restrict__ Abf,
                                                      const __bf16* __restrict__ Xin,
                                                      __bf16* __restrict__ xg,
                                                      int Icols, int KIP) {
  __shared__ alignas(16) __bf16 As[32 * 32];    // row-major: [n_local][k=m_local]
  __shared__ alignas(16) __bf16 XsT[128 * 32];  // transposed: [i][m_local]
  const int tid  = threadIdx.x;
  const int wave = __builtin_amdgcn_readfirstlane(tid >> 5);  // scalar -> s_cbranch
  const int lane = tid & 31;
  const int n0   = blockIdx.x * 32;
  const int b    = blockIdx.y;
  const int numOT = (Icols + 15) >> 4;
  const __bf16* Xt = Xin + (unsigned)b * Icols * NN;  // [i][m]

  v8f acc[2][2] = {};

  const int la = tid * 8, ra = la >> 5, ca = la & 31;  // A staging chunk
  const int nXvec = (Icols * 32) >> 3;                 // 8-elem chunks holding rows i < Icols

  // zero the padded rows (i >= Icols) once: loop-invariant across k-steps
  for (int v = nXvec + tid; v < 512; v += 128) {
    const uint4 z = {0u, 0u, 0u, 0u};
    *(uint4*)(XsT + v * 8) = z;
  }

  for (int m0 = 0; m0 < NN; m0 += 32) {
    // stage A tile 32x32 row-major: one async 16B per thread
    async_copy16(As + la, Abf + (unsigned)(n0 + ra) * NN + (m0 + ca));
    // stage X^T tile (rows i < Icols): async 16B chunks
    for (int v = tid; v < nXvec; v += 128) {
      const int l = v * 8;
      async_copy16(XsT + l, Xt + (unsigned)(l >> 5) * NN + (m0 + (l & 31)));
    }
    async_wait_all();
    __syncthreads();

    const v16bf a0 = lds_afrag(As, 32);
    const v16bf a1 = lds_afrag(As + 16 * 32, 32);
#pragma unroll
    for (int slot = 0; slot < 2; ++slot) {
      const int ot = wave + slot * 4;
      if (ot < numOT) {   // scalar branch: EXEC stays full for WMMA
        const v16bf bf = lds_bfragT(XsT + ot * 16 * 32);
        acc[0][slot] = wmma_bf16(a0, bf, acc[0][slot]);
        acc[1][slot] = wmma_bf16(a1, bf, acc[1][slot]);
      }
    }
    __syncthreads();
  }

  // epilogue: scatter bf16 into xg[(n)*B + b][Icols + i] (32-bit offsets)
  const int nc = lane & 15, mg = lane >> 4;
#pragma unroll
  for (int slot = 0; slot < 2; ++slot) {
    const int ot = wave + slot * 4;
    if (ot >= numOT) continue;
#pragma unroll
    for (int mt = 0; mt < 2; ++mt) {
#pragma unroll
      for (int j = 0; j < 8; ++j) {
        const int mrow = mt * 16 + mg * 8 + j;
        const int i = ot * 16 + nc;
        if (i < Icols)
          xg[((unsigned)(n0 + mrow) * BB + b) * (unsigned)KIP + (Icols + i)] =
              (__bf16)acc[mt][slot][j];
      }
    }
  }
}

// ---------------- GEMM2: per-node  out[b,o] = xg[n](32xKIP) @ Wn[n]^T(OxKIP) + bias, fused GRU ----
// mode 0 (gate): zr[b][n][o] = sigmoid(out)        (O==128)
// mode 1 (upd):  h[b][n][o]  = r*h + (1-r)*tanh(out), r = zr[...,64+o]   (O==64)
__global__ __launch_bounds__(128) void agcrn_gemm_node(const __bf16* __restrict__ xg,
                                                       const __bf16* __restrict__ W,
                                                       const float* __restrict__ bias,
                                                       float* __restrict__ zr,
                                                       float* __restrict__ h,
                                                       int KIP, int O, int mode) {
  __shared__ alignas(16) __bf16 As[32 * 32];    // [b][j_local]
  __shared__ alignas(16) __bf16 WsT[128 * 32];  // [o][j_local]
  const int tid  = threadIdx.x;
  const int wave = __builtin_amdgcn_readfirstlane(tid >> 5);
  const int lane = tid & 31;
  const int n    = blockIdx.x;
  const int numOT = O >> 4;
  const __bf16* xgn = xg + (unsigned)n * BB * KIP;
  const __bf16* Wnn = W + (unsigned)n * O * KIP;   // [o][j]

  v8f acc[2][2] = {};

  const int la = tid * 8, ra = la >> 5, ca = la & 31;
  const int nWvec = (O * 32) >> 3;   // 512 (O=128) or 256 (O=64)

  for (int ks = 0; ks < KIP; ks += 32) {
    // stage xg tile 32(b) x 32(j) row-major: one async 16B per thread
    async_copy16(As + la, xgn + (unsigned)ra * KIP + (ks + ca));
    // stage W^T tile O x 32 (rows o): async 16B chunks
    for (int v = tid; v < nWvec; v += 128) {
      const int l = v * 8;
      async_copy16(WsT + l, Wnn + (unsigned)(l >> 5) * KIP + (ks + (l & 31)));
    }
    async_wait_all();
    __syncthreads();

    const v16bf a0 = lds_afrag(As, 32);
    const v16bf a1 = lds_afrag(As + 16 * 32, 32);
#pragma unroll
    for (int slot = 0; slot < 2; ++slot) {
      const int ot = wave + slot * 4;
      if (ot < numOT) {
        const v16bf bf = lds_bfragT(WsT + ot * 16 * 32);
        acc[0][slot] = wmma_bf16(a0, bf, acc[0][slot]);
        acc[1][slot] = wmma_bf16(a1, bf, acc[1][slot]);
      }
    }
    __syncthreads();
  }

  const int nc = lane & 15, mg = lane >> 4;
#pragma unroll
  for (int slot = 0; slot < 2; ++slot) {
    const int ot = wave + slot * 4;
    if (ot >= numOT) continue;
#pragma unroll
    for (int mt = 0; mt < 2; ++mt) {
#pragma unroll
      for (int j = 0; j < 8; ++j) {
        const int bi = mt * 16 + mg * 8 + j;   // batch index
        const int o  = ot * 16 + nc;
        float val = acc[mt][slot][j] + bias[(unsigned)n * O + o];
        if (mode == 0) {
          zr[((unsigned)bi * NN + n) * 128u + o] = 1.0f / (1.0f + __expf(-val));
        } else {
          const float r = zr[((unsigned)bi * NN + n) * 128u + 64 + o];
          const unsigned hidx = ((unsigned)bi * NN + n) * (unsigned)HH + o;
          const float hc = tanhf(val);
          h[hidx] = r * h[hidx] + (1.0f - r) * hc;
        }
      }
    }
  }
}

// ---------------- output conv: out[b][o][n] = h2[b][n][:] . conv_w[o][:] + conv_b[o] ----------------
__global__ __launch_bounds__(256) void agcrn_conv(const float* __restrict__ h2,
                                                  const float* __restrict__ cw,
                                                  const float* __restrict__ cb,
                                                  float* __restrict__ out) {
  const int idx = blockIdx.x * 256 + threadIdx.x;
  if (idx >= BB * NN) return;
  const int nn = idx & (NN - 1);
  const int b  = idx >> 10;
  const float* hp = h2 + (unsigned)idx * HH;
  float hv[HH];
#pragma unroll
  for (int j = 0; j < HH; ++j) hv[j] = hp[j];
#pragma unroll
  for (int o = 0; o < OUTC; ++o) {
    float a = cb[o];
#pragma unroll
    for (int j = 0; j < HH; ++j) a += hv[j] * cw[o * HH + j];
    out[(unsigned)(b * OUTC + o) * NN + nn] = a;
  }
}

// ---------------- host orchestration ----------------
extern "C" void kernel_launch(void* const* d_in, const int* in_sizes, int n_in,
                              void* d_out, int out_size, void* d_ws, size_t ws_size,
                              hipStream_t stream) {
  const float* source = (const float*)d_in[0];
  const float* emb    = (const float*)d_in[1];
  const float* gw0    = (const float*)d_in[2];
  const float* gb0    = (const float*)d_in[3];
  const float* uw0    = (const float*)d_in[4];
  const float* ub0    = (const float*)d_in[5];
  const float* gw1    = (const float*)d_in[6];
  const float* gb1    = (const float*)d_in[7];
  const float* uw1    = (const float*)d_in[8];
  const float* ub1    = (const float*)d_in[9];
  const float* cw     = (const float*)d_in[10];
  const float* cb     = (const float*)d_in[11];
  float* out = (float*)d_out;

  size_t off = 0;
  auto alloc = [&](size_t bytes) -> void* {
    void* p = (char*)d_ws + off;
    off += (bytes + 255) & ~(size_t)255;
    return p;
  };

  const int KIP0 = 160, KIP1 = 256, I0 = 66, I1 = 128;

  __bf16* Abf = (__bf16*)alloc((size_t)NN * NN * 2);
  __bf16* Wg0 = (__bf16*)alloc((size_t)NN * KIP0 * 128 * 2);
  __bf16* Wu0 = (__bf16*)alloc((size_t)NN * KIP0 * 64 * 2);
  __bf16* Wg1 = (__bf16*)alloc((size_t)NN * KIP1 * 128 * 2);
  __bf16* Wu1 = (__bf16*)alloc((size_t)NN * KIP1 * 64 * 2);
  float*  Bg0 = (float*)alloc((size_t)NN * 128 * 4);
  float*  Bu0 = (float*)alloc((size_t)NN * 64 * 4);
  float*  Bg1 = (float*)alloc((size_t)NN * 128 * 4);
  float*  Bu1 = (float*)alloc((size_t)NN * 64 * 4);
  __bf16* xgb = (__bf16*)alloc((size_t)NN * BB * KIP1 * 2);
  __bf16* agg = (__bf16*)alloc((size_t)BB * I1 * NN * 2);
  float*  zr  = (float*)alloc((size_t)BB * NN * 128 * 4);
  float*  h1  = (float*)alloc((size_t)BB * NN * HH * 4);
  float*  h2  = (float*)alloc((size_t)BB * NN * HH * 4);

  // zero recurrent state + xg (so padded K columns are exact zeros)
  (void)hipMemsetAsync(h1, 0, (size_t)BB * NN * HH * 4, stream);
  (void)hipMemsetAsync(h2, 0, (size_t)BB * NN * HH * 4, stream);
  (void)hipMemsetAsync(xgb, 0, (size_t)NN * BB * KIP1 * 2, stream);

  // one-time: adjacency + per-node weights/biases (time-invariant -> L2-resident on MI455X)
  agcrn_build_A<<<NN, 256, 0, stream>>>(emb, Abf);
  agcrn_build_w<<<dim3((KIP0 * 128 + 255) / 256, NN), 256, 0, stream>>>(emb, gw0, Wg0, I0, KIP0, 128);
  agcrn_build_w<<<dim3((KIP0 * 64  + 255) / 256, NN), 256, 0, stream>>>(emb, uw0, Wu0, I0, KIP0, 64);
  agcrn_build_w<<<dim3((KIP1 * 128 + 255) / 256, NN), 256, 0, stream>>>(emb, gw1, Wg1, I1, KIP1, 128);
  agcrn_build_w<<<dim3((KIP1 * 64  + 255) / 256, NN), 256, 0, stream>>>(emb, uw1, Wu1, I1, KIP1, 64);
  agcrn_build_b<<<NN, 128, 0, stream>>>(emb, gb0, Bg0, 128);
  agcrn_build_b<<<NN, 64,  0, stream>>>(emb, ub0, Bu0, 64);
  agcrn_build_b<<<NN, 128, 0, stream>>>(emb, gb1, Bg1, 128);
  agcrn_build_b<<<NN, 64,  0, stream>>>(emb, ub1, Bu1, 64);

  const dim3 gridAgg(NN / 32, BB);

  for (int t = 0; t < TT; ++t) {
    const float* xt = source + (size_t)t * NN * 2;  // [b][m][c], batch stride T*N*C
    // ---- layer 0, gate ----
    agcrn_pack<<<BB * NN, 128, 0, stream>>>(xt, TT * NN * 2, 2, 2, h1, nullptr, agg, xgb, I0, KIP0);
    agcrn_gemm_agg<<<gridAgg, 128, 0, stream>>>(Abf, agg, xgb, I0, KIP0);
    agcrn_gemm_node<<<NN, 128, 0, stream>>>(xgb, Wg0, Bg0, zr, nullptr, KIP0, 128, 0);
    // ---- layer 0, candidate + GRU update ----
    agcrn_pack<<<BB * NN, 128, 0, stream>>>(xt, TT * NN * 2, 2, 2, h1, zr, agg, xgb, I0, KIP0);
    agcrn_gemm_agg<<<gridAgg, 128, 0, stream>>>(Abf, agg, xgb, I0, KIP0);
    agcrn_gemm_node<<<NN, 128, 0, stream>>>(xgb, Wu0, Bu0, zr, h1, KIP0, 64, 1);
    // ---- layer 1, gate (x_t = h1) ----
    agcrn_pack<<<BB * NN, 128, 0, stream>>>(h1, NN * HH, HH, HH, h2, nullptr, agg, xgb, I1, KIP1);
    agcrn_gemm_agg<<<gridAgg, 128, 0, stream>>>(Abf, agg, xgb, I1, KIP1);
    agcrn_gemm_node<<<NN, 128, 0, stream>>>(xgb, Wg1, Bg1, zr, nullptr, KIP1, 128, 0);
    // ---- layer 1, candidate + GRU update ----
    agcrn_pack<<<BB * NN, 128, 0, stream>>>(h1, NN * HH, HH, HH, h2, zr, agg, xgb, I1, KIP1);
    agcrn_gemm_agg<<<gridAgg, 128, 0, stream>>>(Abf, agg, xgb, I1, KIP1);
    agcrn_gemm_node<<<NN, 128, 0, stream>>>(xgb, Wu1, Bu1, zr, h2, KIP1, 64, 1);
  }

  agcrn_conv<<<(BB * NN + 255) / 256, 256, 0, stream>>>(h2, cw, cb, out);

  (void)in_sizes; (void)n_in; (void)out_size; (void)ws_size;
}
